// LSTM_2078764172298
// MI455X (gfx1250) — matrix operand order repaired and checked by reading, not compiled, not run on hardware
//
#include <hip/hip_runtime.h>
#include <hip/hip_bf16.h>
#include <math.h>

// ---------------------------------------------------------------------------
// LSTM (T=16384, I=256, H=1024, O=512) for MI455X / gfx1250 (wave32, WMMA)
//
// Phase 1: x_gates = x @ W_ih^T + b  -- bf16 WMMA GEMM, hi/lo split for
//          ~f32 accuracy (3x v_wmma_f32_16x16x32_bf16 per K-step).
//          Unroll capped at 2 to stay spill-free.
// Phase 2: persistent 64-WG scan; W_hh staged into LDS (256KB/WGP) via
//          TENSOR_LOAD_TO_LDS (TDM) + s_wait_tensorcnt, c kept WG-local,
//          h double-buffered in global, one monotonic grid barrier per step,
//          xg prefetched one step ahead (global_prefetch_b8).
// Phase 3: FC head + log_softmax, single block.
// ---------------------------------------------------------------------------

typedef __bf16 v16bf __attribute__((ext_vector_type(16)));
typedef float  v8f   __attribute__((ext_vector_type(8)));
typedef unsigned int u32x4 __attribute__((ext_vector_type(4)));
typedef int          i32x8 __attribute__((ext_vector_type(8)));
typedef int          i32x4 __attribute__((ext_vector_type(4)));

#define TSTEPS 16384
#define IDIM   256
#define HDIM   1024
#define G4H    4096          // 4*H gate rows
#define ODIM   512
#define NWG    64            // workgroups in the scan (each owns 16 h-elems)
#define SLICE  16            // h elements per WG
#define WROWS  64            // gate rows per WG (4 gates * SLICE)

__device__ __forceinline__ float sigmoidf_(float x) {
  return 1.0f / (1.0f + expf(-x));
}

// ---------------------------------------------------------------------------
// Phase 1: x_gates[t, n] = sum_k x[t,k] * W_ih[n,k] + (b_ih[n]+b_hh[n])
// One wave per 16x16 output tile; K=256 in 8 steps of 32.
// A-frag (16x32 bf16): lane<16 holds contiguous K {0..15} of row m,
//                                  lane>=16 holds K {16..31}.
// B-frag (32x16 bf16): lane L holds column n = L&15, contiguous K
//                      {0..15} (L<16) or {16..31} (L>=16).
// D (16x16 f32): VGPR v -> row v + 8*(lane>=16), col = lane&15.
// ---------------------------------------------------------------------------
__global__ void xgate_gemm(const float* __restrict__ x,
                           const float* __restrict__ wih,
                           const float* __restrict__ bih,
                           const float* __restrict__ bhh,
                           float* __restrict__ xg) {
  const int wave   = (int)((blockIdx.x * blockDim.x + threadIdx.x) >> 5);
  const int lane   = threadIdx.x & 31;
  const int tile_n = wave % (G4H / 16);
  const int tile_m = wave / (G4H / 16);
  const int half   = lane >> 4;       // 0 or 1
  const int l15    = lane & 15;
  const int m      = tile_m * 16 + l15;   // A row this lane carries
  const int n      = tile_n * 16 + l15;   // B column (= W_ih row) this lane carries
  const float* __restrict__ arow = x   + (size_t)m * IDIM;
  const float* __restrict__ brow = wih + (size_t)n * IDIM;

  v8f acc = {};
#pragma unroll 2
  for (int kk = 0; kk < IDIM; kk += 32) {
    v16bf a_hi, a_lo, b_hi, b_lo;
#pragma unroll
    for (int e = 0; e < 8; ++e) {
      float f0 = arow[kk + half * 16 + e];          // K in {0..7}/{16..23}
      __bf16 h0 = (__bf16)f0;
      a_hi[e] = h0;
      a_lo[e] = (__bf16)(f0 - (float)h0);
      float f1 = arow[kk + half * 16 + 8 + e];      // K in {8..15}/{24..31}
      __bf16 h1 = (__bf16)f1;
      a_hi[8 + e] = h1;
      a_lo[8 + e] = (__bf16)(f1 - (float)h1);
    }
#pragma unroll
    for (int e = 0; e < 16; ++e) {
      float f = brow[kk + half * 16 + e];           // K {0..15}/{16..31}
      __bf16 h = (__bf16)f;
      b_hi[e] = h;
      b_lo[e] = (__bf16)(f - (float)h);
    }
    // f32-accurate product via bf16 split: hi*hi + hi*lo + lo*hi
    acc = __builtin_amdgcn_wmma_f32_16x16x32_bf16(false, a_hi, false, b_hi,
                                                  (short)0, acc, false, false);
    acc = __builtin_amdgcn_wmma_f32_16x16x32_bf16(false, a_hi, false, b_lo,
                                                  (short)0, acc, false, false);
    acc = __builtin_amdgcn_wmma_f32_16x16x32_bf16(false, a_lo, false, b_hi,
                                                  (short)0, acc, false, false);
  }

  const float bias = bih[n] + bhh[n];
#pragma unroll
  for (int v = 0; v < 8; ++v) {
    int row = tile_m * 16 + v + half * 8;
    xg[(size_t)row * G4H + n] = acc[v] + bias;
  }
}

// ---------------------------------------------------------------------------
// TDM: 1-D tile load Global -> LDS (nElems f32, both sides contiguous).
// D# per CDNA5 ISA (normal mode, no pad/iterate).
// ---------------------------------------------------------------------------
__device__ __forceinline__ void tdm_load_1d(const void* gsrc, unsigned lds_off,
                                            unsigned nElems) {
  unsigned long long ga = (unsigned long long)(uintptr_t)gsrc;
  u32x4 g0;
  g0[0] = 1u;                                    // count=1, user mode
  g0[1] = lds_off;                               // lds_addr (bytes)
  g0[2] = (unsigned)(ga & 0xFFFFFFFFull);        // global_addr[31:0]
  g0[3] = (unsigned)((ga >> 32) & 0x01FFFFFFull) // global_addr[56:32]
          | (2u << 30);                          // type=2 ("image")
  i32x8 g1;
  g1[0] = (int)(2u << 16);                       // data_size=2 -> 4 bytes
  g1[1] = (int)((nElems & 0xFFFFu) << 16);       // tensor_dim0[15:0] @63:48
  g1[2] = (int)(((nElems >> 16) & 0xFFFFu)       // tensor_dim0[31:16]
                | (1u << 16));                   // tensor_dim1 = 1
  g1[3] = (int)((nElems & 0xFFFFu) << 16);       // tile_dim0 @127:112
  g1[4] = 1;                                     // tile_dim1 = 1
  g1[5] = (int)nElems;                           // tensor_dim0_stride[31:0]
  g1[6] = (int)((nElems & 0xFFFFu) << 16);       // dim1_stride[15:0] @223:208
  g1[7] = 0;
  i32x4 gz4 = {0, 0, 0, 0};
#if defined(__clang_major__) && (__clang_major__ >= 23)
  i32x8 gz8 = {0, 0, 0, 0, 0, 0, 0, 0};
  __builtin_amdgcn_tensor_load_to_lds(g0, g1, gz4, gz4, gz8, 0);
#else
  __builtin_amdgcn_tensor_load_to_lds(g0, g1, gz4, gz4, 0);
#endif
}

// ---------------------------------------------------------------------------
// Grid-wide monotonic barrier (counter zeroed by init kernel each launch).
// ---------------------------------------------------------------------------
__device__ __forceinline__ void grid_barrier(unsigned* bar, unsigned target) {
  __threadfence();
  __syncthreads();
  if (threadIdx.x == 0) {
    __hip_atomic_fetch_add(bar, 1u, __ATOMIC_ACQ_REL, __HIP_MEMORY_SCOPE_AGENT);
    while (__hip_atomic_load(bar, __ATOMIC_ACQUIRE, __HIP_MEMORY_SCOPE_AGENT) <
           target) {
      __builtin_amdgcn_s_sleep(2);
    }
  }
  __syncthreads();
}

__global__ void init_ws(unsigned* bar) {
  if (threadIdx.x == 0) *bar = 0u;
}

// ---------------------------------------------------------------------------
// Phase 2: persistent scan. WG b owns h[b*16 .. b*16+15] and the 64 gate rows
// feeding them. Those rows are 4 contiguous 64KB blocks of W_hh (16 rows per
// gate) -> staged into LDS with 4 TDM tile loads, resident for all steps.
// Dot products: wave w handles rows w*8..w*8+7; lanes stride k (bank-conflict
// free, no padding needed); wave-local __shfl_xor reduction.
// ---------------------------------------------------------------------------
__global__ void lstm_scan(const float* __restrict__ xg,
                          const float* __restrict__ h0,
                          const float* __restrict__ c0,
                          const float* __restrict__ whh,
                          float* __restrict__ hbufs,  // 2*HDIM (double buffer)
                          float* __restrict__ cfin,   // HDIM
                          unsigned* __restrict__ bar) {
  extern __shared__ float smem[];
  float* Wl   = smem;                   // WROWS * HDIM (256KB, contiguous)
  float* hbuf = Wl + WROWS * HDIM;      // HDIM
  float* xgl  = hbuf + HDIM;            // WROWS
  float* gate = xgl + WROWS;            // WROWS
  float* cv   = gate + WROWS;           // SLICE

  const int b   = blockIdx.x;
  const int tid = threadIdx.x;

  // Stage W_hh slice via Tensor Data Mover: for gate g, global rows
  // [g*HDIM + b*16, +16) x HDIM cols are contiguous (64KB) -> LDS row block
  // [g*16, +16) which is also contiguous.
  if (tid < 32) {                        // wave 0 issues the 4 descriptors
#pragma unroll
    for (int g = 0; g < 4; ++g) {
      const float* src = whh + ((size_t)g * HDIM + (size_t)b * SLICE) * HDIM;
      unsigned lds_off = (unsigned)(size_t)(Wl + g * SLICE * HDIM);
      tdm_load_1d(src, lds_off, SLICE * HDIM);   // 16384 f32 = 64KB
    }
    __builtin_amdgcn_s_wait_tensorcnt(0);
  }
  if (tid < SLICE) {
    cv[tid] = c0[b * SLICE + tid];
    hbufs[b * SLICE + tid] = h0[b * SLICE + tid];   // buffer 0 = initial h
  }
  __syncthreads();                       // LDS (W) visible to all waves

  unsigned epoch = 0;
  grid_barrier(bar, ++epoch * NWG);      // all WGs staged + h0 published

  const int wv = tid >> 5;               // wave 0..7, owns rows wv*8..wv*8+7
  const int ln = tid & 31;
  const int gaterow = (tid >> 4) * HDIM + b * SLICE + (tid & 15); // tid<64

  for (int t = 0; t < TSTEPS; ++t) {
    const float* hcur = hbufs + (size_t)(t & 1) * HDIM;
    float* hnext      = hbufs + (size_t)((t + 1) & 1) * HDIM;

    for (int idx = tid; idx < HDIM; idx += 256) hbuf[idx] = hcur[idx];
    if (tid < WROWS) {
      xgl[tid] = xg[(size_t)t * G4H + gaterow];
      if (t + 1 < TSTEPS)
        __builtin_prefetch(&xg[(size_t)(t + 1) * G4H + gaterow], 0, 1);
    }
    __syncthreads();

    // Hoist h into registers: lane ln holds h[j*32+ln], j=0..31.
    float hr[32];
#pragma unroll
    for (int j = 0; j < 32; ++j) hr[j] = hbuf[j * 32 + ln];

#pragma unroll
    for (int i = 0; i < 8; ++i) {
      const int rid = wv * 8 + i;
      const float* wr = Wl + rid * HDIM;
      float s = 0.0f;
#pragma unroll
      for (int j = 0; j < 32; ++j) s = fmaf(wr[j * 32 + ln], hr[j], s);
#pragma unroll
      for (int off = 16; off > 0; off >>= 1) s += __shfl_xor(s, off, 32);
      if (ln == 0) gate[rid] = s + xgl[rid];
    }
    __syncthreads();

    if (tid < SLICE) {
      float gi = gate[tid],      gf = gate[16 + tid];
      float gg = gate[32 + tid], go = gate[48 + tid];
      float cn = sigmoidf_(gf) * cv[tid] + sigmoidf_(gi) * tanhf(gg);
      cv[tid] = cn;
      hnext[b * SLICE + tid] = sigmoidf_(go) * tanhf(cn);
    }
    grid_barrier(bar, ++epoch * NWG);
  }

  if (tid < SLICE) cfin[b * SLICE + tid] = cv[tid];
}

// ---------------------------------------------------------------------------
// Phase 3: logits = hn @ fc_w^T + fc_b ; out = [log_softmax(logits), hn, cn]
// ---------------------------------------------------------------------------
__global__ void head_kernel(const float* __restrict__ hfin,
                            const float* __restrict__ cfin,
                            const float* __restrict__ fw,
                            const float* __restrict__ fb,
                            float* __restrict__ out) {
  __shared__ float hv[HDIM];
  __shared__ float lg[ODIM];
  __shared__ float red[256];
  const int tid = threadIdx.x;

  for (int i = tid; i < HDIM; i += 256) hv[i] = hfin[i];
  __syncthreads();

  for (int o = tid; o < ODIM; o += 256) {
    const float* w = fw + (size_t)o * HDIM;
    float s = fb[o];
#pragma unroll 8
    for (int k = 0; k < HDIM; ++k) s = fmaf(w[k], hv[k], s);
    lg[o] = s;
  }
  __syncthreads();

  red[tid] = fmaxf(lg[tid], lg[tid + 256]);
  __syncthreads();
  for (int w = 128; w > 0; w >>= 1) {
    if (tid < w) red[tid] = fmaxf(red[tid], red[tid + w]);
    __syncthreads();
  }
  float mx = red[0];
  __syncthreads();

  red[tid] = expf(lg[tid] - mx) + expf(lg[tid + 256] - mx);
  __syncthreads();
  for (int w = 128; w > 0; w >>= 1) {
    if (tid < w) red[tid] += red[tid + w];
    __syncthreads();
  }
  float lse = mx + logf(red[0]);

  for (int o = tid; o < ODIM; o += 256) out[o] = lg[o] - lse;
  for (int i = tid; i < HDIM; i += 256) {
    out[ODIM + i]        = hfin[i];
    out[ODIM + HDIM + i] = cfin[i];
  }
}

// ---------------------------------------------------------------------------
extern "C" void kernel_launch(void* const* d_in, const int* in_sizes, int n_in,
                              void* d_out, int out_size, void* d_ws,
                              size_t ws_size, hipStream_t stream) {
  const float* x   = (const float*)d_in[0];
  const float* h0  = (const float*)d_in[1];
  const float* c0  = (const float*)d_in[2];
  const float* wih = (const float*)d_in[3];
  const float* whh = (const float*)d_in[4];
  const float* bih = (const float*)d_in[5];
  const float* bhh = (const float*)d_in[6];
  const float* fcw = (const float*)d_in[7];
  const float* fcb = (const float*)d_in[8];
  float* out = (float*)d_out;

  // Workspace layout (floats): xg | h double-buffer | c_final | pad | barrier
  float* xg    = (float*)d_ws;                          // TSTEPS*G4H
  float* hbufs = xg + (size_t)TSTEPS * G4H;             // 2*HDIM
  float* cfin  = hbufs + 2 * HDIM;                      // HDIM
  unsigned* bar = (unsigned*)(cfin + HDIM + 64);        // own cacheline

  init_ws<<<1, 32, 0, stream>>>(bar);

  {
    int waves  = (TSTEPS / 16) * (G4H / 16);            // 262144 tiles
    int blocks = waves / 8;                              // 8 waves / block
    xgate_gemm<<<blocks, 256, 0, stream>>>(x, wih, bih, bhh, xg);
  }

  size_t smem =
      (size_t)(WROWS * HDIM + HDIM + WROWS + WROWS + SLICE) * sizeof(float);
  lstm_scan<<<NWG, 256, smem, stream>>>(xg, h0, c0, whh, hbufs, cfin, bar);

  head_kernel<<<1, 256, 0, stream>>>(hbufs + (size_t)(TSTEPS & 1) * HDIM, cfin,
                                     fcw, fcb, out);
}
